// TinyMultiHeadAttention_8065948582388
// MI455X (gfx1250) — matrix operand order repaired and checked by reading, not compiled, not run on hardware
//
#include <hip/hip_runtime.h>
#include <hip/hip_bf16.h>

// ---------------------------------------------------------------------------
// TinyMultiHeadAttention forward for MI455X (gfx1250, wave32, WMMA).
// f32->f16 convert, QKV gemm (64x16 per wave, B-fragment reuse x4),
// flash attention (online softmax, logits stay in VGPRs), out-proj gemm.
// ---------------------------------------------------------------------------

typedef __attribute__((ext_vector_type(16))) _Float16 v16h;
typedef __attribute__((ext_vector_type(8)))  _Float16 v8h;
typedef __attribute__((ext_vector_type(8)))  float    v8f;

#define DEV __device__ __forceinline__

namespace cfg {
constexpr int Bd = 2, S = 2048, D = 768, H = 12, HD = 64;
constexpr int M  = Bd * S;      // 4096 rows of x
constexpr int TN = D / 16;      // 48 column tiles
constexpr int RT = 4;           // row tiles (of 16) per wave in the GEMMs
// workspace layout (bytes)
constexpr size_t XH  = 0;
constexpr size_t WQ  = XH + (size_t)M * D * 2;
constexpr size_t WK  = WQ + (size_t)D * D * 2;
constexpr size_t WV  = WK + (size_t)D * D * 2;
constexpr size_t WO  = WV + (size_t)D * D * 2;
constexpr size_t QO  = WO + (size_t)D * D * 2;
constexpr size_t KO  = QO + (size_t)Bd * H * S * HD * 2;
constexpr size_t VO  = KO + (size_t)Bd * H * S * HD * 2;
constexpr size_t CTX = VO + (size_t)Bd * H * S * HD * 2;
}

// --- WMMA fragment loaders (ISA 7.12.2 VGPR layouts, wave32) ---------------
// A (16x32 f16, MxK): lane l holds row m=l%16; k = (e&7) + (e>>3)*16 + (l>>4)*8
DEV v16h load_frag_a(const _Float16* __restrict__ base, int ld, int m_lane, int hi) {
  const _Float16* p = base + (size_t)m_lane * ld + hi * 8;
  union { v16h v; v8h h[2]; } u;
  u.h[0] = *(const v8h*)(p);
  u.h[1] = *(const v8h*)(p + 16);
  return u.v;
}
// B (32x16 f16, KxN): lane l holds col n=l%16; k = (e&7) + (e>>3)*16 + (l>>4)*8
// `base` is row n of an n-major, k-contiguous operand.
DEV v16h load_frag_b(const _Float16* __restrict__ base, int ld, int n_lane, int hi) {
  const _Float16* p = base + (size_t)n_lane * ld + hi * 8;
  union { v16h v; v8h h[2]; } u;
  u.h[0] = *(const v8h*)(p);
  u.h[1] = *(const v8h*)(p + 16);
  return u.v;
}
DEV v8f wmma_f16(v16h a, v16h b, v8f c) {
  return __builtin_amdgcn_wmma_f32_16x16x32_f16(false, a, false, b, (short)0, c,
                                                false, false);
}

// --- precision conversion --------------------------------------------------
__global__ void __launch_bounds__(256)
k_cvt_f16(const float* __restrict__ in, _Float16* __restrict__ out, int n) {
  int i = blockIdx.x * 256 + threadIdx.x;
  if (i < n) out[i] = (_Float16)in[i];
}
// 32x32 LDS-tiled transpose, coalesced on both global sides.
__global__ void __launch_bounds__(256)
k_transpose_f16(const float* __restrict__ in, _Float16* __restrict__ out, int dim) {
  __shared__ float tile[32][33];
  int nt = dim / 32;
  int bx = blockIdx.x % nt, by = blockIdx.x / nt;
  int tx = threadIdx.x & 31, ty = threadIdx.x >> 5;  // 32 x 8
#pragma unroll
  for (int r = 0; r < 32; r += 8)
    tile[ty + r][tx] = in[(size_t)(by * 32 + ty + r) * dim + bx * 32 + tx];
  __syncthreads();
#pragma unroll
  for (int r = 0; r < 32; r += 8)
    out[(size_t)(bx * 32 + ty + r) * dim + by * 32 + tx] =
        (_Float16)tile[tx][ty + r];
}

// --- fused QKV projection: 64x16 output per wave, B fragments reused x4 ----
// Q -> [B,H,S,64] (pre-scaled by 1/sqrt(HD)), K -> [B,H,S,64], V -> [B,H,64,S]
__global__ void __launch_bounds__(256)
k_gemm_qkv(const _Float16* __restrict__ xh,
           const _Float16* __restrict__ wqt, const _Float16* __restrict__ wkt,
           const _Float16* __restrict__ wvt,
           const float* __restrict__ bq, const float* __restrict__ bk,
           const float* __restrict__ bv,
           _Float16* __restrict__ Q, _Float16* __restrict__ K,
           _Float16* __restrict__ Vt) {
  using namespace cfg;
  int lane = threadIdx.x & 31;
  int wave = blockIdx.x * 8 + (threadIdx.x >> 5);
  int hi = lane >> 4, nidx = lane & 15, rbase = hi * 8;
  int row0 = (wave / TN) * (16 * RT);
  int n0   = (wave % TN) * 16;

  v8f accq[RT], acck[RT], accv[RT];
#pragma unroll
  for (int t = 0; t < RT; ++t) { accq[t] = {}; acck[t] = {}; accv[t] = {}; }

  for (int k0 = 0; k0 < D; k0 += 32) {
    v16h bQ = load_frag_b(wqt + (size_t)n0 * D + k0, D, nidx, hi);
    v16h bK = load_frag_b(wkt + (size_t)n0 * D + k0, D, nidx, hi);
    v16h bV = load_frag_b(wvt + (size_t)n0 * D + k0, D, nidx, hi);
#pragma unroll
    for (int t = 0; t < RT; ++t) {
      v16h a = load_frag_a(xh + (size_t)(row0 + 16 * t) * D + k0, D, nidx, hi);
      accq[t] = wmma_f16(a, bQ, accq[t]);
      acck[t] = wmma_f16(a, bK, acck[t]);
      accv[t] = wmma_f16(a, bV, accv[t]);
    }
  }
  int h = n0 >> 6, d0 = n0 & 63;
  int col = n0 + nidx;
  float bqv = bq[col], bkv = bk[col], bvv = bv[col];
  const float scale = 0.125f;  // 1/sqrt(64)
#pragma unroll
  for (int t = 0; t < RT; ++t) {
    int row_t = row0 + 16 * t;
    int b = row_t / S;                 // 64-row tile never crosses a batch
    int sbase = row_t - b * S;
#pragma unroll
    for (int e = 0; e < 8; ++e) {
      int srow = sbase + rbase + e;
      size_t qk_idx = (((size_t)b * H + h) * S + srow) * HD + d0 + nidx;
      Q[qk_idx] = (_Float16)((accq[t][e] + bqv) * scale);
      K[qk_idx] = (_Float16)(acck[t][e] + bkv);
      Vt[(((size_t)b * H + h) * HD + d0 + nidx) * S + srow] =
          (_Float16)(accv[t][e] + bvv);
    }
  }
}

// --- flash attention: per-wave 16-query tile, 32-key steps -----------------
__global__ void __launch_bounds__(256)
k_flash(const _Float16* __restrict__ Q, const _Float16* __restrict__ K,
        const _Float16* __restrict__ Vt, _Float16* __restrict__ ctx) {
  using namespace cfg;
  __shared__ __align__(16) _Float16 ldsP[8][16][32];  // per-wave P transpose
  int lane = threadIdx.x & 31;
  int wv = threadIdx.x >> 5;
  int hi = lane >> 4, nidx = lane & 15, rbase = hi * 8;
  constexpr int QT = S / 128;  // 16 query tiles of 128 per (b,h)
  int bh = blockIdx.x / QT;
  int q0 = (blockIdx.x % QT) * 128 + wv * 16;
  const _Float16* Qb = Q + (size_t)bh * S * HD;
  const _Float16* Kb = K + (size_t)bh * S * HD;
  const _Float16* Vb = Vt + (size_t)bh * HD * S;

  v16h aq0 = load_frag_a(Qb + (size_t)q0 * HD, HD, nidx, hi);
  v16h aq1 = load_frag_a(Qb + (size_t)q0 * HD + 32, HD, nidx, hi);

  v8f acc0 = {}, acc1 = {}, acc2 = {}, acc3 = {};
  float mrow[8], lrow[8];
#pragma unroll
  for (int e = 0; e < 8; ++e) { mrow[e] = -3.0e38f; lrow[e] = 0.f; }

  for (int j0 = 0; j0 <= q0 + 15; j0 += 32) {
    if (j0 + 32 <= q0 + 15) {  // prefetch next K/V tiles (global_prefetch_b8)
      __builtin_prefetch(Kb + (size_t)(j0 + 32) * HD, 0, 1);
      __builtin_prefetch(Vb + j0 + 32, 0, 1);
    }
    // S tile: rows = queries, cols = keys (two 16-wide key groups)
    v16h bk00 = load_frag_b(Kb + (size_t)j0 * HD, HD, nidx, hi);
    v16h bk01 = load_frag_b(Kb + (size_t)j0 * HD + 32, HD, nidx, hi);
    v16h bk10 = load_frag_b(Kb + (size_t)(j0 + 16) * HD, HD, nidx, hi);
    v16h bk11 = load_frag_b(Kb + (size_t)(j0 + 16) * HD + 32, HD, nidx, hi);
    v8f z = {};
    v8f s0 = wmma_f16(aq0, bk00, z);
    s0 = wmma_f16(aq1, bk01, s0);
    v8f s1 = wmma_f16(aq0, bk10, z);
    s1 = wmma_f16(aq1, bk11, s1);

    bool need_mask = (j0 + 31 > q0);
    float p0[8], p1[8], cm[8];
#pragma unroll
    for (int e = 0; e < 8; ++e) {
      float a = s0[e], bvl = s1[e];
      if (need_mask) {
        int qi = q0 + rbase + e;
        if (j0 + nidx > qi)      a   = -1e9f;
        if (j0 + 16 + nidx > qi) bvl = -1e9f;
      }
      p0[e] = a; p1[e] = bvl;
      cm[e] = fmaxf(a, bvl);
    }
#pragma unroll
    for (int msk = 1; msk < 16; msk <<= 1)
#pragma unroll
      for (int e = 0; e < 8; ++e)
        cm[e] = fmaxf(cm[e], __shfl_xor(cm[e], msk, 32));
    float rs[8], corr[8];
#pragma unroll
    for (int e = 0; e < 8; ++e) {
      float mnew = fmaxf(mrow[e], cm[e]);
      corr[e] = __expf(mrow[e] - mnew);
      p0[e] = __expf(p0[e] - mnew);
      p1[e] = __expf(p1[e] - mnew);
      rs[e] = p0[e] + p1[e];
      mrow[e] = mnew;
    }
#pragma unroll
    for (int msk = 1; msk < 16; msk <<= 1)
#pragma unroll
      for (int e = 0; e < 8; ++e)
        rs[e] += __shfl_xor(rs[e], msk, 32);
#pragma unroll
    for (int e = 0; e < 8; ++e) {
      lrow[e] = lrow[e] * corr[e] + rs[e];
      acc0[e] *= corr[e]; acc1[e] *= corr[e];
      acc2[e] *= corr[e]; acc3[e] *= corr[e];
      // C-layout -> A-layout transpose via per-wave LDS tile (no barrier:
      // each wave owns its buffer; DS ops are in-order within a wave).
      ldsP[wv][rbase + e][nidx]      = (_Float16)p0[e];
      ldsP[wv][rbase + e][16 + nidx] = (_Float16)p1[e];
    }
    v16h ap = load_frag_a(&ldsP[wv][0][0], 32, nidx, hi);
    acc0 = wmma_f16(ap, load_frag_b(Vb + j0,          S, nidx, hi), acc0);
    acc1 = wmma_f16(ap, load_frag_b(Vb + 16 * S + j0, S, nidx, hi), acc1);
    acc2 = wmma_f16(ap, load_frag_b(Vb + 32 * S + j0, S, nidx, hi), acc2);
    acc3 = wmma_f16(ap, load_frag_b(Vb + 48 * S + j0, S, nidx, hi), acc3);
  }
  int b = bh / H, h = bh - b * H;
#pragma unroll
  for (int e = 0; e < 8; ++e) {
    float inv = 1.0f / lrow[e];
    int srow = q0 + rbase + e;
    _Float16* o = ctx + ((size_t)b * S + srow) * D + h * HD;
    o[nidx]      = (_Float16)(acc0[e] * inv);
    o[16 + nidx] = (_Float16)(acc1[e] * inv);
    o[32 + nidx] = (_Float16)(acc2[e] * inv);
    o[48 + nidx] = (_Float16)(acc3[e] * inv);
  }
}

// --- output projection: 64x16 per wave, ctx(f16) @ Wo + bo -> f32 ----------
__global__ void __launch_bounds__(256)
k_gemm_out(const _Float16* __restrict__ ah, const _Float16* __restrict__ wot,
           const float* __restrict__ bo, float* __restrict__ out) {
  using namespace cfg;
  int lane = threadIdx.x & 31;
  int wave = blockIdx.x * 8 + (threadIdx.x >> 5);
  int hi = lane >> 4, nidx = lane & 15, rbase = hi * 8;
  int row0 = (wave / TN) * (16 * RT);
  int n0   = (wave % TN) * 16;

  v8f acc[RT];
#pragma unroll
  for (int t = 0; t < RT; ++t) acc[t] = {};
  for (int k0 = 0; k0 < D; k0 += 32) {
    v16h b = load_frag_b(wot + (size_t)n0 * D + k0, D, nidx, hi);
#pragma unroll
    for (int t = 0; t < RT; ++t) {
      v16h a = load_frag_a(ah + (size_t)(row0 + 16 * t) * D + k0, D, nidx, hi);
      acc[t] = wmma_f16(a, b, acc[t]);
    }
  }
  float bov = bo[n0 + nidx];
#pragma unroll
  for (int t = 0; t < RT; ++t)
#pragma unroll
    for (int e = 0; e < 8; ++e)
      out[(size_t)(row0 + 16 * t + rbase + e) * D + n0 + nidx] =
          acc[t][e] + bov;
}

// ---------------------------------------------------------------------------
extern "C" void kernel_launch(void* const* d_in, const int* in_sizes, int n_in,
                              void* d_out, int out_size, void* d_ws,
                              size_t ws_size, hipStream_t stream) {
  using namespace cfg;
  (void)in_sizes; (void)n_in; (void)out_size; (void)ws_size;
  const float* x  = (const float*)d_in[0];
  // d_in[1] = mask: exactly 1-tril(S,S); applied analytically in k_flash.
  const float* Wq = (const float*)d_in[2];
  const float* bq = (const float*)d_in[3];
  const float* Wk = (const float*)d_in[4];
  const float* bk = (const float*)d_in[5];
  const float* Wv = (const float*)d_in[6];
  const float* bv = (const float*)d_in[7];
  const float* Wo = (const float*)d_in[8];
  const float* bo = (const float*)d_in[9];
  float* out = (float*)d_out;

  char* ws = (char*)d_ws;
  _Float16* xh  = (_Float16*)(ws + XH);
  _Float16* wqt = (_Float16*)(ws + WQ);
  _Float16* wkt = (_Float16*)(ws + WK);
  _Float16* wvt = (_Float16*)(ws + WV);
  _Float16* wot = (_Float16*)(ws + WO);
  _Float16* Qh  = (_Float16*)(ws + QO);
  _Float16* Kh  = (_Float16*)(ws + KO);
  _Float16* Vth = (_Float16*)(ws + VO);
  _Float16* ctx = (_Float16*)(ws + CTX);

  const int n_x = M * D;                    // 3,145,728
  const int tr_blocks = (D / 32) * (D / 32);  // 576
  k_cvt_f16<<<(n_x + 255) / 256, 256, 0, stream>>>(x, xh, n_x);
  k_transpose_f16<<<tr_blocks, 256, 0, stream>>>(Wq, wqt, D);
  k_transpose_f16<<<tr_blocks, 256, 0, stream>>>(Wk, wkt, D);
  k_transpose_f16<<<tr_blocks, 256, 0, stream>>>(Wv, wvt, D);
  k_transpose_f16<<<tr_blocks, 256, 0, stream>>>(Wo, wot, D);

  const int gemm_blocks = (M / (16 * RT)) * TN / 8;  // 384 blocks of 8 waves
  k_gemm_qkv<<<gemm_blocks, 256, 0, stream>>>(xh, wqt, wkt, wvt, bq, bk, bv,
                                              Qh, Kh, Vth);
  k_flash<<<Bd * H * (S / 128), 256, 0, stream>>>(Qh, Kh, Vth, ctx);
  k_gemm_out<<<gemm_blocks, 256, 0, stream>>>(ctx, wot, bo, out);
}
